// SABlock_9148280341107
// MI455X (gfx1250) — matrix operand order repaired and checked by reading, not compiled, not run on hardware
//
#include <hip/hip_runtime.h>
#include <hip/hip_bf16.h>
#include <math.h>

// ---------------------------------------------------------------------------
// Types
// ---------------------------------------------------------------------------
typedef __bf16 bf16;
typedef __bf16 v8bf  __attribute__((ext_vector_type(8)));
typedef __bf16 v16bf __attribute__((ext_vector_type(16)));
typedef float  v8f   __attribute__((ext_vector_type(8)));
typedef unsigned int u32x4 __attribute__((ext_vector_type(4)));
typedef int i32x4 __attribute__((ext_vector_type(4)));
typedef int i32x8 __attribute__((ext_vector_type(8)));

#define SHUF16(lo, hi) __builtin_shufflevector(lo, hi, 0,1,2,3,4,5,6,7,8,9,10,11,12,13,14,15)

// TDM availability (device pass only; host pass sees fallback)
#if defined(__gfx1250__) && __has_builtin(__builtin_amdgcn_tensor_load_to_lds)
#define HAVE_TDM 1
#endif

// Problem constants
static const int kB    = 16;
static const int kC    = 384;
static const int kHW   = 1024;           // 32*32
static const int kTok  = kB * kHW;       // 16384
static const int kHeads = 8;
static const int kDh   = 48;             // padded to 64 inside attention
static const int k3C   = 3 * kC;         // 1152
static const int k4C   = 4 * kC;         // 1536

// ---------------------------------------------------------------------------
// 1) Depthwise 3x3 conv positional encoding + bias + residual, NCHW -> (B,N,C)
// ---------------------------------------------------------------------------
__global__ __launch_bounds__(256) void conv_pe_kernel(
    const float* __restrict__ x, const float* __restrict__ pw,
    const float* __restrict__ pb, float* __restrict__ t1)
{
    int idx = blockIdx.x * 256 + threadIdx.x;      // total B*C*H*W
    int xx = idx & 31;
    int yy = (idx >> 5) & 31;
    int tmp = idx >> 10;
    int c = tmp % kC;
    int b = tmp / kC;
    const float* xp = x + ((size_t)(b * kC + c) << 10);
    const float* wp = pw + c * 9;
    float acc = xp[(yy << 5) + xx] + pb[c];
#pragma unroll
    for (int dy = -1; dy <= 1; dy++) {
#pragma unroll
        for (int dx = -1; dx <= 1; dx++) {
            int y2 = yy + dy, x2 = xx + dx;
            if (y2 >= 0 && y2 < 32 && x2 >= 0 && x2 < 32)
                acc += wp[(dy + 1) * 3 + (dx + 1)] * xp[(y2 << 5) + x2];
        }
    }
    t1[((size_t)b * kHW + (yy << 5) + xx) * kC + c] = acc;
}

// ---------------------------------------------------------------------------
// 2) LayerNorm over C=384 per token; one wave per token; output bf16
// ---------------------------------------------------------------------------
__global__ __launch_bounds__(256) void ln_bf16_kernel(
    const float* __restrict__ in, const float* __restrict__ g,
    const float* __restrict__ bt, bf16* __restrict__ out)
{
    int wave = threadIdx.x >> 5;
    int lane = threadIdx.x & 31;
    int tok  = blockIdx.x * 8 + wave;
    const float* row = in + (size_t)tok * kC;
    float v[12];
    float s = 0.f;
#pragma unroll
    for (int i = 0; i < 12; i++) { v[i] = row[lane + i * 32]; s += v[i]; }
#pragma unroll
    for (int off = 1; off < 32; off <<= 1) s += __shfl_xor(s, off, 32);
    float mu = s * (1.0f / kC);
    float q = 0.f;
#pragma unroll
    for (int i = 0; i < 12; i++) { float d = v[i] - mu; q += d * d; }
#pragma unroll
    for (int off = 1; off < 32; off <<= 1) q += __shfl_xor(q, off, 32);
    float rstd = rsqrtf(q * (1.0f / kC) + 1e-5f);
    bf16* orow = out + (size_t)tok * kC;
#pragma unroll
    for (int i = 0; i < 12; i++) {
        int c = lane + i * 32;
        orow[c] = (bf16)((v[i] - mu) * rstd * g[c] + bt[c]);
    }
}

// ---------------------------------------------------------------------------
// 3) f32 (KxN) -> bf16 transposed (NxK) weight conversion
// ---------------------------------------------------------------------------
__global__ __launch_bounds__(256) void cvt_bf16_t_kernel(
    const float* __restrict__ in, bf16* __restrict__ out, int K, int N)
{
    int idx = blockIdx.x * 256 + threadIdx.x;     // K*N total, k fastest on out
    if (idx >= K * N) return;
    int k = idx % K;
    int n = idx / K;
    out[(size_t)n * K + k] = (bf16)in[(size_t)k * N + n];
}

// ---------------------------------------------------------------------------
// TDM helper: DMA a rows x 32 bf16 tile (row stride = strideElems) into LDS,
// with LDS padding 64B-row + 16B pad => 40-element LDS row stride.
// D# per CDNA5 ISA ch.8. Issued by one wave; tracked via TENSORcnt.
// ---------------------------------------------------------------------------
#if defined(HAVE_TDM)
__device__ __forceinline__ void tdm_load_tile32(
    const bf16* gsrc, unsigned ldsByte, int rows, int strideElems)
{
    unsigned long long ga = (unsigned long long)(uintptr_t)gsrc;
    u32x4 g0;
    g0[0] = 1u;                                            // count=1, user mode
    g0[1] = ldsByte;                                       // lds_addr
    g0[2] = (unsigned)(ga & 0xFFFFFFFFu);                  // global_addr[31:0]
    g0[3] = (unsigned)((ga >> 32) & 0x01FFFFFFu) | (2u << 30); // addr[56:32], type=2
    i32x8 g1;
    g1[0] = (int)((1u << 16)     // data_size = 1 (2 bytes)
                | (1u << 20)     // pad_enable
                | (3u << 22)     // pad_interval: 16 DWORDs (64B = one tile row)
                | (3u << 25));   // pad_amount: 4 DWORDs (16B = 8 elements)
    g1[1] = (int)(32u << 16);                              // tensor_dim0 = 32
    g1[2] = (int)((unsigned)rows << 16);                   // tensor_dim1 = rows
    g1[3] = (int)(32u << 16);                              // tile_dim0 = 32
    g1[4] = rows;                                          // tile_dim1 = rows
    g1[5] = strideElems;                                   // tensor_dim0_stride
    g1[6] = 0;
    g1[7] = 0;
    i32x4 z4 = {0, 0, 0, 0};
#if __clang_major__ >= 23
    i32x8 z8 = {0, 0, 0, 0, 0, 0, 0, 0};
    __builtin_amdgcn_tensor_load_to_lds(g0, g1, z4, z4, z8, 0);
#else
    __builtin_amdgcn_tensor_load_to_lds(g0, g1, z4, z4, 0);
#endif
}

// s_wait_tensorcnt requires an immediate -> dedicated wrappers
__device__ __forceinline__ void tdm_wait0()
{
#if __has_builtin(__builtin_amdgcn_s_wait_tensorcnt)
    __builtin_amdgcn_s_wait_tensorcnt(0);
#else
    asm volatile("s_wait_tensorcnt 0x0" ::: "memory");
#endif
}
__device__ __forceinline__ void tdm_wait2()
{
#if __has_builtin(__builtin_amdgcn_s_wait_tensorcnt)
    __builtin_amdgcn_s_wait_tensorcnt(2);
#else
    asm volatile("s_wait_tensorcnt 0x2" ::: "memory");
#endif
}
#endif // HAVE_TDM

// ---------------------------------------------------------------------------
// Shared WMMA micro-step: one 16-row strip (wave) x 64 cols x K=32 slab.
// A-frag: lanes 0-15 -> K{0..7,16..23}; lanes 16-31 -> K{8..15,24..31}.
// B-frag: lanes 0-15 -> K{0..7,16..23}; lanes 16-31 -> K{8..15,24..31}.
// ---------------------------------------------------------------------------
__device__ __forceinline__ void mma_step(
    const bf16 (*__restrict__ As)[40], const bf16 (*__restrict__ Bs)[40],
    int w, int half, int l16, v8f acc[4])
{
    const int arow = (w << 4) + l16;
    const int ak0  = half ? 8 : 0;
    v8bf alo = *reinterpret_cast<const v8bf*>(&As[arow][ak0]);
    v8bf ahi = *reinterpret_cast<const v8bf*>(&As[arow][ak0 + 16]);
    v16bf af = SHUF16(alo, ahi);
#pragma unroll
    for (int nt = 0; nt < 4; nt++) {
        const int col = (nt << 4) + l16;
        const int bk0 = half << 3;
        v8bf blo = *reinterpret_cast<const v8bf*>(&Bs[col][bk0]);
        v8bf bhi = *reinterpret_cast<const v8bf*>(&Bs[col][bk0 + 16]);
        v16bf bf = SHUF16(blo, bhi);
        acc[nt] = __builtin_amdgcn_wmma_f32_16x16x32_bf16(
            false, af, false, bf, (short)0, acc[nt], false, false);
    }
}

// ---------------------------------------------------------------------------
// 4) Tiled WMMA GEMM: C(MxN) = A(MxK,bf16) * Bt(NxK,bf16)^T + epilogue.
//    BM=128 BN=64 BK=32; 256 threads = 8 waves; TDM double-buffered staging.
// ---------------------------------------------------------------------------
enum { EPI_BIAS = 0, EPI_BIAS_RES = 1, EPI_GELU_BF16 = 2 };

template <int EPI>
__global__ __launch_bounds__(256, 2) void gemm_bf16_kernel(
    const bf16* __restrict__ A, const bf16* __restrict__ Bt,
    const float* __restrict__ bias, const float* __restrict__ res,
    float* __restrict__ Cf, bf16* __restrict__ Cb,
    int M, int N, int K)
{
    __shared__ bf16 As[2][128][40];   // row stride 80B (16B-aligned)
    __shared__ bf16 Bs[2][64][40];    // stored [n][k]

    const int tid  = threadIdx.x;
    const int w    = tid >> 5;
    const int lane = tid & 31;
    const int half = lane >> 4;
    const int l16  = lane & 15;
    const int m0   = blockIdx.y * 128;
    const int n0   = blockIdx.x * 64;
    const int nk   = K >> 5;

    v8f acc[4];
#pragma unroll
    for (int nt = 0; nt < 4; nt++)
#pragma unroll
        for (int r = 0; r < 8; r++) acc[nt][r] = 0.f;

    const bf16* gA = A  + (size_t)m0 * K;
    const bf16* gB = Bt + (size_t)n0 * K;

#if defined(HAVE_TDM)
    // -------- Tensor-Data-Mover double-buffered pipeline --------
    if (w == 0) {
        tdm_load_tile32(gA, (unsigned)(uintptr_t)&As[0][0][0], 128, K);
        tdm_load_tile32(gB, (unsigned)(uintptr_t)&Bs[0][0][0], 64, K);
    }
    int buf = 0;
    for (int kt = 0; kt < nk; kt++) {
        if (w == 0) {
            if (kt + 1 < nk) {
                tdm_load_tile32(gA + (kt + 1) * 32,
                                (unsigned)(uintptr_t)&As[buf ^ 1][0][0], 128, K);
                tdm_load_tile32(gB + (kt + 1) * 32,
                                (unsigned)(uintptr_t)&Bs[buf ^ 1][0][0], 64, K);
                tdm_wait2();     // tiles for step kt have landed
            } else {
                tdm_wait0();
            }
        }
        __syncthreads();
        mma_step(As[buf], Bs[buf], w, half, l16, acc);
        __syncthreads();         // everyone done with buf before TDM reuses it
        buf ^= 1;
    }
#else
    // -------- Fallback: manual vector staging (single buffer) --------
    const int arow_l = tid >> 1;
    const int akc    = (tid & 1) * 16;
    const int brow   = tid >> 2;
    const int bkc    = (tid & 3) * 8;
    for (int kt = 0; kt < nk; kt++) {
        const int k0 = kt << 5;
        const bf16* ga = gA + (size_t)arow_l * K + k0 + akc;
        *reinterpret_cast<uint4*>(&As[0][arow_l][akc])     = reinterpret_cast<const uint4*>(ga)[0];
        *reinterpret_cast<uint4*>(&As[0][arow_l][akc + 8]) = reinterpret_cast<const uint4*>(ga)[1];
        const bf16* gb = gB + (size_t)brow * K + k0 + bkc;
        *reinterpret_cast<v8bf*>(&Bs[0][brow][bkc]) = *reinterpret_cast<const v8bf*>(gb);
        __syncthreads();
        if (kt + 1 < nk) {
            __builtin_prefetch(ga + 32, 0, 1);
            __builtin_prefetch(gb + 32, 0, 1);
        }
        mma_step(As[0], Bs[0], w, half, l16, acc);
        __syncthreads();
    }
#endif

    // epilogue; C layout: vgpr r, lanes0-15 -> (M=r,N=l16); lanes16-31 -> (M=r+8)
#pragma unroll
    for (int nt = 0; nt < 4; nt++) {
#pragma unroll
        for (int r = 0; r < 8; r++) {
            int m = m0 + (w << 4) + (half << 3) + r;
            int n = n0 + (nt << 4) + l16;
            float v = acc[nt][r] + bias[n];
            if (EPI == EPI_BIAS_RES) {
                v += res[(size_t)m * N + n];
                Cf[(size_t)m * N + n] = v;
            } else if (EPI == EPI_GELU_BF16) {
                float gl = 0.5f * v * (1.0f + erff(v * 0.70710678118f));
                Cb[(size_t)m * N + n] = (bf16)gl;
            } else {
                Cf[(size_t)m * N + n] = v;
            }
        }
    }
}

// ---------------------------------------------------------------------------
// 5) Flash attention: grid (qtiles=8, H=8, B=16); 256 threads; q-tile 128,
//    key chunks of 64, Dh 48 padded to 64. Scale folded into Q staging.
// ---------------------------------------------------------------------------
__global__ __launch_bounds__(256) void attn_kernel(
    const float* __restrict__ qkv, bf16* __restrict__ obf)
{
    const int qb = blockIdx.x * 128;
    const int h  = blockIdx.y;
    const int b  = blockIdx.z;

    __shared__ bf16 Qs[128][72];
    __shared__ bf16 Ks[64][72];
    __shared__ bf16 VTs[64][72];      // [d][key]
    __shared__ bf16 Ps[8][16][72];    // per-wave P staging

    const int tid  = threadIdx.x;
    const int w    = tid >> 5;
    const int lane = tid & 31;
    const int half = lane >> 4;
    const int l16  = lane & 15;
    const float scale = 0.14433756729740643f;   // 1/sqrt(48)

    {   // stage Q (scaled, zero-padded d=48..63)
        int row = tid >> 1;
        int dc  = (tid & 1) * 32;
        const float* src = qkv + (size_t)(b * kHW + qb + row) * k3C + h * kDh;
#pragma unroll
        for (int i = 0; i < 32; i++) {
            int d = dc + i;
            Qs[row][d] = (bf16)((d < kDh) ? src[d] * scale : 0.f);
        }
    }

    float rmax[8], rsum[8];
    v8f oacc[4];
#pragma unroll
    for (int r = 0; r < 8; r++) { rmax[r] = -3.0e38f; rsum[r] = 0.f; }
#pragma unroll
    for (int nt = 0; nt < 4; nt++)
#pragma unroll
        for (int r = 0; r < 8; r++) oacc[nt][r] = 0.f;

    for (int kt = 0; kt < kHW; kt += 64) {
        __syncthreads();
        {   // stage K and V(transposed) chunk
            int key = tid >> 2;
            int dc  = (tid & 3) * 16;
            const float* ksrc = qkv + (size_t)(b * kHW + kt + key) * k3C + kC + h * kDh;
            const float* vsrc = ksrc + kC;
#pragma unroll
            for (int i = 0; i < 16; i++) {
                int d = dc + i;
                Ks[key][d]  = (bf16)((d < kDh) ? ksrc[d] : 0.f);
                VTs[d][key] = (bf16)((d < kDh) ? vsrc[d] : 0.f);
            }
        }
        __syncthreads();

        // S = Q * K^T  (K-dim = 64 -> two x32 WMMA steps)
        v8f s[4];
#pragma unroll
        for (int nt = 0; nt < 4; nt++)
#pragma unroll
            for (int r = 0; r < 8; r++) s[nt][r] = 0.f;
#pragma unroll
        for (int ks = 0; ks < 2; ks++) {
            const int arow = (w << 4) + l16;
            const int ak0  = ks * 32 + (half ? 8 : 0);
            v8bf alo = *reinterpret_cast<const v8bf*>(&Qs[arow][ak0]);
            v8bf ahi = *reinterpret_cast<const v8bf*>(&Qs[arow][ak0 + 16]);
            v16bf af = SHUF16(alo, ahi);
#pragma unroll
            for (int nt = 0; nt < 4; nt++) {
                const int col = (nt << 4) + l16;
                const int bk0 = ks * 32 + (half << 3);
                v8bf blo = *reinterpret_cast<const v8bf*>(&Ks[col][bk0]);
                v8bf bhi = *reinterpret_cast<const v8bf*>(&Ks[col][bk0 + 16]);
                v16bf bf = SHUF16(blo, bhi);
                s[nt] = __builtin_amdgcn_wmma_f32_16x16x32_bf16(
                    false, af, false, bf, (short)0, s[nt], false, false);
            }
        }

        // online softmax (row lives in one 16-lane half)
#pragma unroll
        for (int r = 0; r < 8; r++) {
            float mx = s[0][r];
#pragma unroll
            for (int nt = 1; nt < 4; nt++) mx = fmaxf(mx, s[nt][r]);
#pragma unroll
            for (int off = 1; off < 16; off <<= 1) mx = fmaxf(mx, __shfl_xor(mx, off, 32));
            float mnew = fmaxf(rmax[r], mx);
            float corr = __expf(rmax[r] - mnew);
            rmax[r] = mnew;
            float ps = 0.f;
#pragma unroll
            for (int nt = 0; nt < 4; nt++) {
                float p = __expf(s[nt][r] - mnew);
                s[nt][r] = p; ps += p;
            }
#pragma unroll
            for (int off = 1; off < 16; off <<= 1) ps += __shfl_xor(ps, off, 32);
            rsum[r] = rsum[r] * corr + ps;
#pragma unroll
            for (int nt = 0; nt < 4; nt++) oacc[nt][r] *= corr;
        }

        // P (C-layout) -> bf16 -> LDS -> A-fragment layout
#pragma unroll
        for (int nt = 0; nt < 4; nt++)
#pragma unroll
            for (int r = 0; r < 8; r++)
                Ps[w][(half << 3) + r][(nt << 4) + l16] = (bf16)s[nt][r];
        __syncthreads();

        // O += P * V
#pragma unroll
        for (int ks = 0; ks < 2; ks++) {
            const int ak0 = ks * 32 + (half ? 8 : 0);
            v8bf alo = *reinterpret_cast<const v8bf*>(&Ps[w][l16][ak0]);
            v8bf ahi = *reinterpret_cast<const v8bf*>(&Ps[w][l16][ak0 + 16]);
            v16bf af = SHUF16(alo, ahi);
#pragma unroll
            for (int nt = 0; nt < 4; nt++) {
                const int col = (nt << 4) + l16;          // output d
                const int bk0 = ks * 32 + (half << 3);    // key
                v8bf blo = *reinterpret_cast<const v8bf*>(&VTs[col][bk0]);
                v8bf bhi = *reinterpret_cast<const v8bf*>(&VTs[col][bk0 + 16]);
                v16bf bf = SHUF16(blo, bhi);
                oacc[nt] = __builtin_amdgcn_wmma_f32_16x16x32_bf16(
                    false, af, false, bf, (short)0, oacc[nt], false, false);
            }
        }
    }

    // normalize and emit (b, q, h, d) for d < 48, as bf16 for the proj GEMM
#pragma unroll
    for (int nt = 0; nt < 4; nt++) {
#pragma unroll
        for (int r = 0; r < 8; r++) {
            int d = (nt << 4) + l16;
            if (d < kDh) {
                int q = qb + (w << 4) + (half << 3) + r;
                obf[(size_t)(b * kHW + q) * kC + h * kDh + d] =
                    (bf16)(oacc[nt][r] / rsum[r]);
            }
        }
    }
}

// ---------------------------------------------------------------------------
// 6) (B,N,C) f32 -> (B,C,H,W) f32 output
// ---------------------------------------------------------------------------
__global__ __launch_bounds__(256) void nchw_out_kernel(
    const float* __restrict__ t, float* __restrict__ out)
{
    int idx = blockIdx.x * 256 + threadIdx.x;   // ((b*C + c)<<10) + hw
    int hw = idx & 1023;
    int tmp = idx >> 10;
    int c = tmp % kC;
    int b = tmp / kC;
    out[idx] = t[((size_t)b * kHW + hw) * kC + c];
}

// ---------------------------------------------------------------------------
// Host launcher
// ---------------------------------------------------------------------------
extern "C" void kernel_launch(void* const* d_in, const int* in_sizes, int n_in,
                              void* d_out, int out_size, void* d_ws, size_t ws_size,
                              hipStream_t stream) {
    const float* x     = (const float*)d_in[0];
    const float* pos_w = (const float*)d_in[1];
    const float* pos_b = (const float*)d_in[2];
    const float* g1    = (const float*)d_in[3];
    const float* b1    = (const float*)d_in[4];
    const float* wqkv  = (const float*)d_in[5];
    const float* bqkv  = (const float*)d_in[6];
    const float* wproj = (const float*)d_in[7];
    const float* bproj = (const float*)d_in[8];
    const float* g2    = (const float*)d_in[9];
    const float* b2    = (const float*)d_in[10];
    const float* w1    = (const float*)d_in[11];
    const float* bf1   = (const float*)d_in[12];
    const float* w2    = (const float*)d_in[13];
    const float* bf2   = (const float*)d_in[14];
    float* out = (float*)d_out;

    char* ws = (char*)d_ws;
    size_t off = 0;
    auto alloc = [&](size_t bytes) { size_t r = off; off += (bytes + 255) & ~(size_t)255; return r; };

    size_t o_t1   = alloc((size_t)kTok * kC * 4);    // conv residual (aliased by t3 later)
    size_t o_qkv  = alloc((size_t)kTok * k3C * 4);   // qkv f32 (aliased by MLP hidden later)
    size_t o_xnb  = alloc((size_t)kTok * kC * 2);    // LN1 bf16
    size_t o_obf  = alloc((size_t)kTok * kC * 2);    // attention out bf16
    size_t o_t2   = alloc((size_t)kTok * kC * 4);    // residual 2 f32
    size_t o_ynb  = alloc((size_t)kTok * kC * 2);    // LN2 bf16
    size_t o_wqkvb = alloc((size_t)kC * k3C * 2);    // all weights stored transposed (N x K)
    size_t o_wprojb= alloc((size_t)kC * kC * 2);
    size_t o_w1b  = alloc((size_t)kC * k4C * 2);
    size_t o_w2b  = alloc((size_t)k4C * kC * 2);

    float* t1    = (float*)(ws + o_t1);
    float* qkv   = (float*)(ws + o_qkv);
    bf16*  xnb   = (bf16*)(ws + o_xnb);
    bf16*  obf   = (bf16*)(ws + o_obf);
    float* t2    = (float*)(ws + o_t2);
    bf16*  ynb   = (bf16*)(ws + o_ynb);
    bf16*  wqkvb = (bf16*)(ws + o_wqkvb);
    bf16*  wprojb= (bf16*)(ws + o_wprojb);
    bf16*  w1b   = (bf16*)(ws + o_w1b);
    bf16*  w2b   = (bf16*)(ws + o_w2b);
    bf16*  hb    = (bf16*)(ws + o_qkv);   // reuse: qkv dead after attention
    float* t3    = (float*)(ws + o_t1);   // reuse: t1 dead after proj residual

    // 1) conv positional encoding + residual -> t1 (B,N,C)
    conv_pe_kernel<<<(kB * kC * kHW) / 256, 256, 0, stream>>>(x, pos_w, pos_b, t1);

    // weight conversions with transpose: (KxN) f32 -> (NxK) bf16
    cvt_bf16_t_kernel<<<(kC * k3C) / 256, 256, 0, stream>>>(wqkv,  wqkvb,  kC, k3C);
    cvt_bf16_t_kernel<<<(kC * kC)  / 256, 256, 0, stream>>>(wproj, wprojb, kC, kC);
    cvt_bf16_t_kernel<<<(kC * k4C) / 256, 256, 0, stream>>>(w1,    w1b,    kC, k4C);
    cvt_bf16_t_kernel<<<(k4C * kC) / 256, 256, 0, stream>>>(w2,    w2b,    k4C, kC);

    // 2) LN1 -> bf16
    ln_bf16_kernel<<<kTok / 8, 256, 0, stream>>>(t1, g1, b1, xnb);

    // 3) QKV GEMM: (16384x384) @ (384x1152) + bqkv -> qkv f32
    gemm_bf16_kernel<EPI_BIAS><<<dim3(k3C / 64, kTok / 128), 256, 0, stream>>>(
        xnb, wqkvb, bqkv, nullptr, qkv, nullptr, kTok, k3C, kC);

    // 4) attention -> obf bf16
    attn_kernel<<<dim3(kHW / 128, kHeads, kB), 256, 0, stream>>>(qkv, obf);

    // 5) proj GEMM + residual(t1): t2 = t1 + obf @ wproj + bproj
    gemm_bf16_kernel<EPI_BIAS_RES><<<dim3(kC / 64, kTok / 128), 256, 0, stream>>>(
        obf, wprojb, bproj, t1, t2, nullptr, kTok, kC, kC);

    // 6) LN2 -> bf16
    ln_bf16_kernel<<<kTok / 8, 256, 0, stream>>>(t2, g2, b2, ynb);

    // 7) MLP1 + exact GELU -> hb bf16 (16384x1536)
    gemm_bf16_kernel<EPI_GELU_BF16><<<dim3(k4C / 64, kTok / 128), 256, 0, stream>>>(
        ynb, w1b, bf1, nullptr, nullptr, hb, kTok, k4C, kC);

    // 8) MLP2 + residual(t2): t3 = t2 + hb @ w2 + bf2
    gemm_bf16_kernel<EPI_BIAS_RES><<<dim3(kC / 64, kTok / 128), 256, 0, stream>>>(
        hb, w2b, bf2, t2, t3, nullptr, kTok, kC, k4C);

    // 9) transpose to NCHW output
    nchw_out_kernel<<<(kB * kC * kHW) / 256, 256, 0, stream>>>(t3, out);
}
